// ResBlk_47433618817332
// MI455X (gfx1250) — matrix operand order repaired and checked by reading, not compiled, not run on hardware
//
#include <hip/hip_runtime.h>

typedef __attribute__((ext_vector_type(16))) __bf16 v16bf;
typedef __attribute__((ext_vector_type(8)))  __bf16 v8bf;
typedef __attribute__((ext_vector_type(8)))  float  v8f;

#define EPSF 1e-12f

static constexpr int N = 32, C = 32, H = 256, W = 256, HP = 128, WP = 128;

// workspace layout (bytes)
static constexpr size_t OFF_XT  = 0;                                   // bf16 NHWC 256^2 (reused for h2)
static constexpr size_t OFF_H1  = OFF_XT  + (size_t)N*H*W*C*2;         // bf16 NHWC 256^2
static constexpr size_t OFF_HPL = OFF_H1  + (size_t)N*H*W*C*2;         // pooled h2, bf16 NHWC 128^2
static constexpr size_t OFF_XP  = OFF_HPL + (size_t)N*HP*WP*C*2;       // pooled x, bf16 NHWC 128^2
static constexpr size_t OFF_CS  = OFF_XP  + (size_t)N*HP*WP*C*2;       // f32 chan-sumsq full res
static constexpr size_t OFF_NRM = OFF_CS  + (size_t)N*H*W*4;           // f32 patch norm full res
static constexpr size_t OFF_CSP = OFF_NRM + (size_t)N*H*W*4;           // f32 chan-sumsq pooled
static constexpr size_t OFF_W1B = OFF_CSP + (size_t)N*HP*WP*4;         // B-image conv1 (18432B)
static constexpr size_t OFF_W2B = OFF_W1B + 32768;
static constexpr size_t OFF_W3B = OFF_W2B + 32768;
static constexpr size_t OFF_PEF = OFF_W3B + 4096;                      // 96 f32
static constexpr size_t OFF_QEF = OFF_PEF + 512;                       // 3 f32

// ---------------------------------------------------------------------------
// gfx1250 async global->LDS copy (16B per lane), tracked by ASYNCcnt.
// ISA: GLOBAL_LOAD_ASYNC_TO_LDS_B128  vdst(LDS byte addr), vaddr(global), off
// ---------------------------------------------------------------------------
__device__ __forceinline__ unsigned lds_addr_of(const void* p) {
  return (unsigned)(uintptr_t)(__attribute__((address_space(3))) const void*)p;
}
__device__ __forceinline__ void async_g2l_b128(const void* g, void* l) {
  unsigned laddr = lds_addr_of(l);
  asm volatile("global_load_async_to_lds_b128 %0, %1, off"
               :: "v"(laddr), "v"(g) : "memory");
}
__device__ __forceinline__ void wait_async0() {
  asm volatile("s_wait_asynccnt 0x0" ::: "memory");
}

// ---------------------------------------------------------------------------
// Prep: per-Cout weight L2 normalization, bake B-matrix register images
// (lane = {Khalf(lane>>4), Ncol(lane&15)}, element e -> K=(lane>>4)*16+e),
// and p_eff = (p/10)^2, q_eff = |q|/100.
// ---------------------------------------------------------------------------
__global__ void prep_kernel(const float* w1, const float* p1, const float* q1,
                            const float* w2, const float* p2, const float* q2,
                            const float* w3, const float* p3, const float* q3,
                            __bf16* b1, __bf16* b2, __bf16* b3,
                            float* peff, float* qeff) {
  __shared__ float invn[3][32];
  int tid = threadIdx.x;
  if (tid < 96) {
    int cv = tid >> 5, o = tid & 31;
    const float* w  = cv == 0 ? w1 : (cv == 1 ? w2 : w3);
    const float* pp = cv == 0 ? p1 : (cv == 1 ? p2 : p3);
    int taps = (cv == 2) ? 1 : 9;
    float s = 0.f;
    for (int i = 0; i < 32 * taps; ++i) { float v = w[o * 32 * taps + i]; s += v * v; }
    invn[cv][o] = __builtin_amdgcn_rsqf(s + EPSF);
    float pv = pp[o] * 0.1f;
    peff[tid] = pv * pv;
  }
  if (tid < 3) {
    const float* qq = tid == 0 ? q1 : (tid == 1 ? q2 : q3);
    qeff[tid] = __builtin_fabsf(qq[0]) * 0.01f;
  }
  __syncthreads();
  for (int cv = 0; cv < 3; ++cv) {
    const float* w = cv == 0 ? w1 : (cv == 1 ? w2 : w3);
    __bf16* bd = cv == 0 ? b1 : (cv == 1 ? b2 : b3);
    int taps = (cv == 2) ? 1 : 9;
    int total = taps * 2 * 32 * 16;
    for (int e = tid; e < total; e += 256) {
      int tap = e >> 10;               // 2*32*16 = 1024 per tap
      int r = e & 1023;
      int tile = r >> 9;               // cout half
      int lane = (r >> 4) & 31;
      int el   = r & 15;
      int k    = ((lane >> 4) << 4) + el;      // cin
      int co   = tile * 16 + (lane & 15);      // cout
      float v = w[(co * 32 + k) * taps + tap] * invn[cv][co];
      bd[e] = (__bf16)v;
    }
  }
}

// ---------------------------------------------------------------------------
// NCHW f32 -> NHWC bf16 (coalesced reads per channel, 4x16B vector stores),
// plus per-pixel channel sum-of-squares.
// ---------------------------------------------------------------------------
__global__ void transpose_kernel(const float* __restrict__ x,
                                 __bf16* __restrict__ xT,
                                 float* __restrict__ cs) {
  int t = blockIdx.x * 256 + threadIdx.x;        // n*H*W + h*W + w
  int w = t & 255, h = (t >> 8) & 255, n = t >> 16;
  float s = 0.f;
  v8bf buf[4];
#pragma unroll
  for (int c = 0; c < 32; ++c) {
    float v = x[(((size_t)n * 32 + c) * 256 + h) * 256 + w];
    s += v * v;
    buf[c >> 3][c & 7] = (__bf16)v;
  }
  v8bf* o = (v8bf*)(xT + (size_t)t * 32);
  o[0] = buf[0]; o[1] = buf[1]; o[2] = buf[2]; o[3] = buf[3];
  cs[t] = s;
}

// 3x3 zero-padded window sum of chan-sumsq, then sqrt(+eps) -> patch norm
__global__ void win3_kernel(const float* __restrict__ cs, float* __restrict__ nrm) {
  int t = blockIdx.x * 256 + threadIdx.x;
  int w = t & 255, h = (t >> 8) & 255, n = t >> 16;
  float s = 0.f;
  for (int dy = -1; dy <= 1; ++dy) {
    int hh = h + dy; if ((unsigned)hh >= 256u) continue;
    for (int dx = -1; dx <= 1; ++dx) {
      int ww = w + dx; if ((unsigned)ww >= 256u) continue;
      s += cs[((size_t)n * 256 + hh) * 256 + ww];
    }
  }
  nrm[t] = __builtin_amdgcn_sqrtf(s + EPSF);
}

// per-pixel channel sum-of-squares of a bf16 NHWC tensor
__global__ void chansq_kernel(const __bf16* __restrict__ in, float* __restrict__ cs, int total) {
  int t = blockIdx.x * 256 + threadIdx.x;
  if (t >= total) return;
  const v8bf* p = (const v8bf*)(in + (size_t)t * 32);
  float s = 0.f;
#pragma unroll
  for (int g = 0; g < 4; ++g) {
    v8bf v = p[g];
#pragma unroll
    for (int i = 0; i < 8; ++i) { float f = (float)v[i]; s += f * f; }
  }
  cs[t] = s;
}

// 2x2/2 max-abs (sign preserving) pool, NHWC bf16 256^2 -> 128^2,
// 8 channels per thread
__global__ void pool_kernel(const __bf16* __restrict__ in, __bf16* __restrict__ out) {
  int t = blockIdx.x * 256 + threadIdx.x;        // (n, hp, wp, g)
  int g = t & 3;
  int wp = (t >> 2) & 127;
  int hp = (t >> 9) & 127;
  int n  = t >> 16;
  size_t ib = (((size_t)n * 256 + 2 * hp) * 256 + 2 * wp) * 32 + g * 8;
  v8bf a = *(const v8bf*)(in + ib);
  v8bf b = *(const v8bf*)(in + ib + 32);
  v8bf c = *(const v8bf*)(in + ib + 256 * 32);
  v8bf d = *(const v8bf*)(in + ib + 256 * 32 + 32);
  v8bf o;
#pragma unroll
  for (int i = 0; i < 8; ++i) {
    float fa = (float)a[i], fb = (float)b[i], fc = (float)c[i], fd = (float)d[i];
    float best = fa;
    if (__builtin_fabsf(fb) > __builtin_fabsf(best)) best = fb;
    if (__builtin_fabsf(fc) > __builtin_fabsf(best)) best = fc;
    if (__builtin_fabsf(fd) > __builtin_fabsf(best)) best = fd;
    o[i] = (__bf16)best;
  }
  *(v8bf*)(out + (((size_t)n * 128 + hp) * 128 + wp) * 32 + g * 8) = o;
}

// ---------------------------------------------------------------------------
// SCS 3x3 conv, pad=1. Block = 8 waves = 8 rows x 64 cols output tile.
// Each wave owns one row: 4 M-tiles (64 pixels) x 32 Cout -> 8 f32 accs.
// Halo (10x66 px, zero-padded) + B-image staged to LDS with
// global_load_async_to_lds_b128 (ASYNCcnt). Per tap: 4 B ds_loads shared by
// 4 M-tiles + 8 A ds_loads feed 8 v_wmma_f32_16x16x32_bf16 (1.5 ds/WMMA),
// with ~110 live VGPRs (no spills).
// ---------------------------------------------------------------------------
__global__ void scs3x3_kernel(const __bf16* __restrict__ xin, const float* __restrict__ nrm,
                              const __bf16* __restrict__ Bimg, const float* __restrict__ peff,
                              const float* __restrict__ qeffp, __bf16* __restrict__ out) {
  __shared__ __align__(16) __bf16 Bs[9 * 2 * 32 * 16];   // 18432 B
  __shared__ __align__(16) __bf16 tile[10 * 66 * 32];    // 42240 B halo tile

  int tid = threadIdx.x;
  int b = blockIdx.x;                    // n(32) x hb(32) x wb(4)
  int wb = b & 3, hb = (b >> 2) & 31, n = b >> 7;
  int h0 = hb * 8, w0 = wb * 64;

  // stage B image (1152 x 16B chunks) asynchronously
  for (int i = tid; i < 1152; i += 256)
    async_g2l_b128(Bimg + i * 8, &Bs[i * 8]);

  // stage zero-padded 10x66 halo: 660 pixels (64B = 4x b128 each)
  for (int pi = tid; pi < 660; pi += 256) {
    int row = pi / 66, col = pi % 66;
    int gh = h0 - 1 + row, gw = w0 - 1 + col;
    __bf16* l = &tile[(size_t)pi * 32];
    if ((unsigned)gh < 256u && (unsigned)gw < 256u) {
      const __bf16* g = xin + (((size_t)n * 256 + gh) * 256 + gw) * 32;
#pragma unroll
      for (int j = 0; j < 4; ++j) async_g2l_b128(g + j * 8, l + j * 8);
    } else {
      v8bf z = {};
      v8bf* lv = (v8bf*)l;
#pragma unroll
      for (int j = 0; j < 4; ++j) lv[j] = z;
    }
  }
  wait_async0();
  __syncthreads();

  int lane = tid & 31, wave = tid >> 5;
  int m = lane & 15, halfK = lane >> 4;

  v8f acc[8];                           // [mtile][couthalf]
#pragma unroll
  for (int i = 0; i < 8; ++i) acc[i] = (v8f){};

#pragma unroll
  for (int t = 0; t < 9; ++t) {
    int ky = t / 3, kx = t % 3;
    v16bf b0 = *(const v16bf*)&Bs[((t * 2 + 0) * 32 + lane) * 16];
    v16bf b1 = *(const v16bf*)&Bs[((t * 2 + 1) * 32 + lane) * 16];
#pragma unroll
    for (int g = 0; g < 4; ++g) {
      const __bf16* ap = &tile[(((wave + ky) * 66) + (g * 16 + m + kx)) * 32 + halfK * 16];
      v8bf lo = *(const v8bf*)ap;          // Cin {half*16..+7}
      v8bf hi = *(const v8bf*)(ap + 8);    // Cin {half*16+8..+15}
      v16bf a = __builtin_shufflevector(lo, hi, 0, 1, 2, 3, 4, 5, 6, 7,
                                                8, 9, 10, 11, 12, 13, 14, 15);
      acc[g * 2 + 0] = __builtin_amdgcn_wmma_f32_16x16x32_bf16(false, a, false, b0, (short)0, acc[g * 2 + 0], false, false);
      acc[g * 2 + 1] = __builtin_amdgcn_wmma_f32_16x16x32_bf16(false, a, false, b1, (short)0, acc[g * 2 + 1], false, false);
    }
  }

  int h = h0 + wave;
  int ncol = lane & 15, halfM = lane >> 4;
  float qe = qeffp[0];
  float pe0 = peff[ncol], pe1 = peff[ncol + 16];
#pragma unroll
  for (int g = 0; g < 4; ++g) {
    v8f acc0 = acc[g * 2 + 0], acc1 = acc[g * 2 + 1];
#pragma unroll
    for (int r = 0; r < 8; ++r) {
      int mm = r + halfM * 8;
      size_t pix = ((size_t)n * 256 + h) * 256 + (w0 + g * 16 + mm);
      float dn = nrm[pix] + qe;
      float c0 = acc0[r] / dn, c1 = acc1[r] / dn;
      float a0 = __builtin_fabsf(c0) + EPSF, a1 = __builtin_fabsf(c1) + EPSF;
      float o0 = __builtin_copysignf(__builtin_amdgcn_exp2f(pe0 * __builtin_amdgcn_logf(a0)), c0);
      float o1 = __builtin_copysignf(__builtin_amdgcn_exp2f(pe1 * __builtin_amdgcn_logf(a1)), c1);
      // lanes 0-15 cover one pixel's cout 0-15 -> contiguous 32B runs
      out[pix * 32 + ncol]      = (__bf16)o0;
      out[pix * 32 + ncol + 16] = (__bf16)o1;
    }
  }
}

// ---------------------------------------------------------------------------
// Final: SCS 1x1 (pad=0) on pooled x, + pooled-h residual, f32 NCHW out.
// Block = one full output row (n, h, w=0..127); results staged in LDS and
// written channel-major so every NCHW cacheline is produced contiguously.
// ---------------------------------------------------------------------------
__global__ void final_kernel(const __bf16* __restrict__ xp, const float* __restrict__ csp,
                             const __bf16* __restrict__ Bimg, const float* __restrict__ peff,
                             const float* __restrict__ qeffp, const __bf16* __restrict__ hpool,
                             float* __restrict__ dout) {
  __shared__ __align__(16) float rowbuf[32 * 128];       // [cout][w], 16 KB

  int tid = threadIdx.x;
  int lane = tid & 31, wave = tid >> 5;
  int b = blockIdx.x;                    // (n, h) : 32*128
  int n = b >> 7, h = b & 127;
  int w0 = wave * 16;
  int m = lane & 15, halfK = lane >> 4;

  // B tiles in registers (2 x v16bf)
  v16bf b0 = *(const v16bf*)(Bimg + ((size_t)0 * 32 + lane) * 16);
  v16bf b1 = *(const v16bf*)(Bimg + ((size_t)1 * 32 + lane) * 16);

  size_t base = (((size_t)n * 128 + h) * 128 + (w0 + m)) * 32;
  v8bf lo = *(const v8bf*)(xp + base + halfK * 16);
  v8bf hi = *(const v8bf*)(xp + base + 8 + halfK * 16);
  v16bf a = __builtin_shufflevector(lo, hi, 0, 1, 2, 3, 4, 5, 6, 7,
                                            8, 9, 10, 11, 12, 13, 14, 15);
  v8f acc0 = {}, acc1 = {};
  acc0 = __builtin_amdgcn_wmma_f32_16x16x32_bf16(false, a, false, b0, (short)0, acc0, false, false);
  acc1 = __builtin_amdgcn_wmma_f32_16x16x32_bf16(false, a, false, b1, (short)0, acc1, false, false);

  float qe = qeffp[0];
  int ncol = lane & 15, halfM = lane >> 4;
  float pe0 = peff[ncol], pe1 = peff[ncol + 16];
#pragma unroll
  for (int r = 0; r < 8; ++r) {
    int mm = r + halfM * 8;
    int wcol = w0 + mm;
    size_t pix = ((size_t)n * 128 + h) * 128 + wcol;
    float dn = __builtin_amdgcn_sqrtf(csp[pix] + EPSF) + qe;
    float c0 = acc0[r] / dn, c1 = acc1[r] / dn;
    float a0 = __builtin_fabsf(c0) + EPSF, a1 = __builtin_fabsf(c1) + EPSF;
    float o0 = __builtin_copysignf(__builtin_amdgcn_exp2f(pe0 * __builtin_amdgcn_logf(a0)), c0);
    float o1 = __builtin_copysignf(__builtin_amdgcn_exp2f(pe1 * __builtin_amdgcn_logf(a1)), c1);
    float h0v = (float)hpool[pix * 32 + ncol];
    float h1v = (float)hpool[pix * 32 + ncol + 16];
    rowbuf[ncol * 128 + wcol]        = o0 + h0v;
    rowbuf[(ncol + 16) * 128 + wcol] = o1 + h1v;
  }
  __syncthreads();

  // thread t writes cout = t>>3, w = (t&7)*16 .. +15 : 64B contiguous
  int cout = tid >> 3, ws = (tid & 7) * 16;
  const float4* src = (const float4*)&rowbuf[cout * 128 + ws];
  float4* dst = (float4*)&dout[(((size_t)n * 32 + cout) * 128 + h) * 128 + ws];
#pragma unroll
  for (int j = 0; j < 4; ++j) dst[j] = src[j];
}

// ---------------------------------------------------------------------------
extern "C" void kernel_launch(void* const* d_in, const int* in_sizes, int n_in,
                              void* d_out, int out_size, void* d_ws, size_t ws_size,
                              hipStream_t stream) {
  const float* x  = (const float*)d_in[0];
  const float* w1 = (const float*)d_in[1];
  const float* p1 = (const float*)d_in[2];
  const float* q1 = (const float*)d_in[3];
  const float* w2 = (const float*)d_in[4];
  const float* p2 = (const float*)d_in[5];
  const float* q2 = (const float*)d_in[6];
  const float* w3 = (const float*)d_in[7];
  const float* p3 = (const float*)d_in[8];
  const float* q3 = (const float*)d_in[9];

  unsigned char* ws = (unsigned char*)d_ws;
  __bf16* xT  = (__bf16*)(ws + OFF_XT);    // also reused for h2
  __bf16* h1  = (__bf16*)(ws + OFF_H1);
  __bf16* hpl = (__bf16*)(ws + OFF_HPL);
  __bf16* xpb = (__bf16*)(ws + OFF_XP);
  float*  cs  = (float*)(ws + OFF_CS);
  float*  nrm = (float*)(ws + OFF_NRM);
  float*  csp = (float*)(ws + OFF_CSP);
  __bf16* b1  = (__bf16*)(ws + OFF_W1B);
  __bf16* b2  = (__bf16*)(ws + OFF_W2B);
  __bf16* b3  = (__bf16*)(ws + OFF_W3B);
  float*  pef = (float*)(ws + OFF_PEF);
  float*  qef = (float*)(ws + OFF_QEF);

  prep_kernel<<<1, 256, 0, stream>>>(w1, p1, q1, w2, p2, q2, w3, p3, q3,
                                     b1, b2, b3, pef, qef);
  transpose_kernel<<<8192, 256, 0, stream>>>(x, xT, cs);
  win3_kernel<<<8192, 256, 0, stream>>>(cs, nrm);
  scs3x3_kernel<<<4096, 256, 0, stream>>>(xT, nrm, b1, pef + 0, qef + 0, h1);
  pool_kernel<<<8192, 256, 0, stream>>>(xT, xpb);                    // y = pool(x)
  chansq_kernel<<<2048, 256, 0, stream>>>(xpb, csp, 32 * 128 * 128); // for 1x1 norm
  chansq_kernel<<<8192, 256, 0, stream>>>(h1, cs, 32 * 256 * 256);
  win3_kernel<<<8192, 256, 0, stream>>>(cs, nrm);
  scs3x3_kernel<<<4096, 256, 0, stream>>>(h1, nrm, b2, pef + 32, qef + 1, xT); // h2 into xT
  pool_kernel<<<8192, 256, 0, stream>>>(xT, hpl);                    // pool(h2)
  final_kernel<<<4096, 256, 0, stream>>>(xpb, csp, b3, pef + 64, qef + 2, hpl,
                                         (float*)d_out);
}
